// MultiheadFlashDiff2_29274497089979
// MI455X (gfx1250) — compile-verified, matrix operand order checked by reading, not run on hardware
//
#include <hip/hip_runtime.h>

// ---------------------------------------------------------------------------
// MultiheadFlashDiff2 forward for MI455X (gfx1250, wave32, WMMA bf16 + TDM)
// ---------------------------------------------------------------------------
typedef __attribute__((ext_vector_type(16))) __bf16       v16bf;
typedef __attribute__((ext_vector_type(8)))  float        v8f;
typedef __attribute__((ext_vector_type(4)))  unsigned int v4u;
typedef __attribute__((ext_vector_type(2)))  unsigned int v2u;
typedef __attribute__((ext_vector_type(4)))  float        v4f;
typedef __attribute__((ext_vector_type(8)))  int          v8i;
typedef __attribute__((ext_vector_type(4)))  int          v4i;

#define EMBED 2048
#define SEQ   2048
#define NHEAD 16
#define HDIM  64
#define LAMBDA_INIT 0.78360576653162442f

// ---- Tensor Data Mover availability / arity detection ----------------------
#if defined(__has_builtin)
#  if __has_builtin(__builtin_amdgcn_tensor_load_to_lds)
#    define HAVE_TDM 1
#  endif
#endif
#ifndef HAVE_TDM
#  define HAVE_TDM 0
#endif
#if __has_include(<hip/amd_detail/amd_gfx1250_TDM.h>)
#  define TDM_SIX_ARGS 1   // therock-10.0 headers -> 6-arg builtin
#else
#  define TDM_SIX_ARGS 0   // ROCm 7.2 clang-22 -> 5-arg builtin
#endif

__device__ __forceinline__ unsigned short f2bf(float f) {
  unsigned int u = __float_as_uint(f);
  return (unsigned short)((u + 0x7FFFu + ((u >> 16) & 1u)) >> 16);
}

union Frag {           // 32 bytes: one bf16 WMMA A/B operand (8 VGPRs)
  v4u   u[2];
  v16bf v;
};

#if HAVE_TDM
// 2D tile load, bf16 elements (data_size=2B), LDS padding per pad_interval /
// pad_amount (D# group1 bits 20,24:22,31:25).  tile_d0 = elems per row,
// tile_d1 = rows, stride0 = row stride in elements.
__device__ __forceinline__ void tdm_load_2d(const void* gaddr, unsigned lds_addr,
                                            unsigned tile_d0, unsigned tile_d1,
                                            unsigned long long stride0,
                                            unsigned pad_interval, unsigned pad_amount) {
  unsigned long long ga = (unsigned long long)(uintptr_t)gaddr;
  v4u g0;
  g0[0] = 1u;                                              // count=1, user mode
  g0[1] = lds_addr;                                        // LDS byte address
  g0[2] = (unsigned)ga;                                    // global_addr[31:0]
  g0[3] = (unsigned)((ga >> 32) & 0x01FFFFFFu) | (2u << 30); // addr[56:32]+type=2
  v8i g1;
  g1[0] = (int)((1u << 16)                                 // data_size = 2 bytes
              | (1u << 20)                                 // pad_enable
              | (pad_interval << 22) | (pad_amount << 25));
  unsigned d0 = 2048u, d1 = 2048u;                         // tensor dims (OOB only)
  g1[1] = (int)((d0 & 0xFFFFu) << 16);                     // tensor_dim0[15:0]
  g1[2] = (int)(((d0 >> 16) & 0xFFFFu) | ((d1 & 0xFFFFu) << 16));
  g1[3] = (int)(((d1 >> 16) & 0xFFFFu) | (tile_d0 << 16)); // tile_dim0
  g1[4] = (int)(tile_d1 & 0xFFFFu);                        // tile_dim1, tile_dim2=0
  g1[5] = (int)(unsigned)(stride0 & 0xFFFFFFFFu);          // tensor_dim0_stride lo
  g1[6] = (int)(unsigned)((stride0 >> 32) & 0xFFFFu);      // stride hi, dim1_stride=0
  g1[7] = 0;
  v4i gz = {0, 0, 0, 0};
#if TDM_SIX_ARGS
  v8i gz8 = {0, 0, 0, 0, 0, 0, 0, 0};
  __builtin_amdgcn_tensor_load_to_lds(g0, g1, gz, gz, gz8, 0);
#else
  __builtin_amdgcn_tensor_load_to_lds(g0, g1, gz, gz, 0);
#endif
}
#endif  // HAVE_TDM

// ---------------------------------------------------------------------------
// 1) f32 -> bf16 convert (vectorized)
// ---------------------------------------------------------------------------
__global__ __launch_bounds__(256) void cvt_bf16_kernel(const float* __restrict__ in,
                                                       unsigned short* __restrict__ out,
                                                       int n) {
  int i = (blockIdx.x * 256 + threadIdx.x) * 4;
  if (i < n) {
    v4f f = *(const v4f*)(in + i);
    v2u o;
    o[0] = (unsigned int)f2bf(f[0]) | ((unsigned int)f2bf(f[1]) << 16);
    o[1] = (unsigned int)f2bf(f[2]) | ((unsigned int)f2bf(f[3]) << 16);
    *(v2u*)(out + i) = o;
  }
}

// ---------------------------------------------------------------------------
// 2) lambda_full = exp(dot(lq1,lk1)) - exp(dot(lq2,lk2)) + LAMBDA_INIT
// ---------------------------------------------------------------------------
__global__ __launch_bounds__(32) void lambda_kernel(const float* __restrict__ lq1,
                                                    const float* __restrict__ lk1,
                                                    const float* __restrict__ lq2,
                                                    const float* __restrict__ lk2,
                                                    float* __restrict__ out) {
  int l = threadIdx.x;
  float d1 = lq1[l] * lk1[l] + lq1[l + 32] * lk1[l + 32];
  float d2 = lq2[l] * lk2[l] + lq2[l + 32] * lk2[l + 32];
#pragma unroll
  for (int m = 16; m > 0; m >>= 1) {
    d1 += __shfl_xor(d1, m);
    d2 += __shfl_xor(d2, m);
  }
  if (l == 0) out[0] = __expf(d1) - __expf(d2) + LAMBDA_INIT;
}

// ---------------------------------------------------------------------------
// 3) C[2048,2048] = A @ W^T (bf16 in, bf16/f32 out)
//    128x128 block tile, 8 waves (4x2), 32-wide K steps.
//    TDM path: wave0 drives tensor_load_to_lds into double-buffered LDS,
//    pipelined with s_wait_tensorcnt; compute waves only do ds_load + wmma.
// ---------------------------------------------------------------------------
__global__ __launch_bounds__(256, 1) void gemm2048_kernel(
    const unsigned short* __restrict__ A, const unsigned short* __restrict__ W,
    unsigned short* __restrict__ outB, float* __restrict__ outF) {
  const int K = EMBED, N = EMBED;
  const int bm = blockIdx.x * 128, bn = blockIdx.y * 128;
  const int tid = threadIdx.x;
  const int w = tid >> 5, lane = tid & 31;
  const int wm = (w >> 1) * 32, wn = (w & 1) * 64;
  const int lo = lane & 15, kh = lane >> 4;

  __shared__ unsigned short sA[2][128][48];  // 16 DW data + 8 DW pad per row
  __shared__ unsigned short sB[2][128][48];

  v8f acc[2][4];
  const v8f vz = {0.f, 0.f, 0.f, 0.f, 0.f, 0.f, 0.f, 0.f};
#pragma unroll
  for (int mi = 0; mi < 2; mi++)
#pragma unroll
    for (int ni = 0; ni < 4; ni++) acc[mi][ni] = vz;

  const int NSTEP = K / 32;

#if HAVE_TDM
  const unsigned ldsA[2] = {(unsigned)(uintptr_t)&sA[0][0][0],
                            (unsigned)(uintptr_t)&sA[1][0][0]};
  const unsigned ldsB[2] = {(unsigned)(uintptr_t)&sB[0][0][0],
                            (unsigned)(uintptr_t)&sB[1][0][0]};
  auto issueStep = [&](int s) {
    int b = s & 1;
    tdm_load_2d(A + (size_t)bm * K + s * 32, ldsA[b], 32, 128, K, 3, 7);
    tdm_load_2d(W + (size_t)bn * K + s * 32, ldsB[b], 32, 128, K, 3, 7);
  };
  if (w == 0) issueStep(0);
#endif

  for (int s = 0; s < NSTEP; s++) {
    const int b = s & 1;
#if HAVE_TDM
    if (w == 0) {
      if (s + 1 < NSTEP) {
        issueStep(s + 1);                                // prefetch next tile
        __builtin_amdgcn_s_wait_tensorcnt((short)2);     // tile s complete
      } else {
        __builtin_amdgcn_s_wait_tensorcnt((short)0);
      }
    }
    __syncthreads();  // tile s visible to all waves
#else
    (void)b;
    __syncthreads();
#pragma unroll
    for (int i = 0; i < 2; i++) {
      int cc = tid + 256 * i;
      int row = cc >> 2, seg = (cc & 3) * 8;
      int k0 = s * 32;
      *(v4u*)&sA[0][row][seg] = *(const v4u*)(A + (size_t)(bm + row) * K + k0 + seg);
      *(v4u*)&sB[0][row][seg] = *(const v4u*)(W + (size_t)(bn + row) * K + k0 + seg);
    }
    if (s + 1 < NSTEP)
      __builtin_prefetch(A + (size_t)(bm + (tid >> 1)) * K + s * 32 + 32, 0, 1);
    __syncthreads();
#endif

    const int rb = HAVE_TDM ? b : 0;
    Frag af[2], bfr[4];
#pragma unroll
    for (int mi = 0; mi < 2; mi++) {
      af[mi].u[0] = *(const v4u*)&sA[rb][wm + mi * 16 + lo][kh * 8];
      af[mi].u[1] = *(const v4u*)&sA[rb][wm + mi * 16 + lo][16 + kh * 8];
    }
#pragma unroll
    for (int ni = 0; ni < 4; ni++) {
      bfr[ni].u[0] = *(const v4u*)&sB[rb][wn + ni * 16 + lo][kh * 8];
      bfr[ni].u[1] = *(const v4u*)&sB[rb][wn + ni * 16 + lo][16 + kh * 8];
    }
#pragma unroll
    for (int mi = 0; mi < 2; mi++)
#pragma unroll
      for (int ni = 0; ni < 4; ni++)
        acc[mi][ni] = __builtin_amdgcn_wmma_f32_16x16x32_bf16(
            false, af[mi].v, false, bfr[ni].v, (short)0, acc[mi][ni], false, false);

#if HAVE_TDM
    __syncthreads();  // reads of buffer b done before TDM reuses it (step s+2)
#endif
  }

  // C/D layout: lane holds col n=lane&15; rows m = (lane>>4)*8 + r
#pragma unroll
  for (int mi = 0; mi < 2; mi++)
#pragma unroll
    for (int ni = 0; ni < 4; ni++)
#pragma unroll
      for (int r = 0; r < 8; r++) {
        int row = bm + wm + mi * 16 + kh * 8 + r;
        int col = bn + wn + ni * 16 + lo;
        float v = acc[mi][ni][r];
        if (outF) outF[(size_t)row * N + col] = v;
        else      outB[(size_t)row * N + col] = f2bf(v);
      }
}

// ---------------------------------------------------------------------------
// 4) Differential flash attention + RMS subln, fused.
//    grid = (head 16, qblock 16); block = 256 threads = 8 waves x 16 q-rows.
//    K tile staged by TDM (wave0), V tile transposed into LDS by all threads.
// ---------------------------------------------------------------------------
__global__ __launch_bounds__(256, 1) void diffattn_kernel(
    const unsigned short* __restrict__ Qb, const unsigned short* __restrict__ Kb,
    const unsigned short* __restrict__ Vb, const float* __restrict__ lam_p,
    const float* __restrict__ subln, unsigned short* __restrict__ attnb) {
  const int h = blockIdx.x;
  const int qb = blockIdx.y;
  const int tid = threadIdx.x;
  const int w = tid >> 5, lane = tid & 31;
  const int lo = lane & 15, kh = lane >> 4;
  const float SC = 0.125f;  // 1/sqrt(64)

  __shared__ unsigned short Kls[32][136];    // 64 DW data + 4 DW pad per row
  __shared__ unsigned short VT[128][48];     // transposed: 128 dims x 32 keys (+pad)
  __shared__ unsigned short Pls[8][16][48];  // per-wave P re-layout scratch

#if HAVE_TDM
  const unsigned ldsK = (unsigned)(uintptr_t)&Kls[0][0];
#endif

  // --- Q fragments (held in VGPRs for whole kernel): [attn a][dim chunk c] ---
  Frag qf[2][2];
  {
    const int qrow = qb * 128 + w * 16 + lo;
    const unsigned short* qptr = Qb + (size_t)qrow * EMBED + h * 128;
#pragma unroll
    for (int a = 0; a < 2; a++)
#pragma unroll
      for (int c = 0; c < 2; c++) {
        int base = a * 64 + c * 32;
        qf[a][c].u[0] = *(const v4u*)(qptr + base + kh * 8);
        qf[a][c].u[1] = *(const v4u*)(qptr + base + 16 + kh * 8);
      }
  }

  float mr[2][8], lr[2][8];
  v8f acc[2][8];
  const v8f vz = {0.f, 0.f, 0.f, 0.f, 0.f, 0.f, 0.f, 0.f};
#pragma unroll
  for (int a = 0; a < 2; a++) {
#pragma unroll
    for (int r = 0; r < 8; r++) { mr[a][r] = -1e30f; lr[a][r] = 0.f; }
#pragma unroll
    for (int d = 0; d < 8; d++) acc[a][d] = vz;
  }

  for (int kt = 0; kt < SEQ / 32; kt++) {
    __syncthreads();  // previous iteration's LDS reads complete
#if HAVE_TDM
    if (w == 0)
      tdm_load_2d(Kb + (size_t)(kt * 32) * EMBED + h * 128, ldsK, 128, 32, EMBED, 5, 3);
#endif
    {  // V tile: load key-major rows, scatter transposed into VT
      int key = tid >> 3, seg = tid & 7;
#if !HAVE_TDM
      const unsigned short* kp = Kb + (size_t)(kt * 32 + key) * EMBED + h * 128 + seg * 16;
      *(v4u*)&Kls[key][seg * 16]     = *(const v4u*)kp;
      *(v4u*)&Kls[key][seg * 16 + 8] = *(const v4u*)(kp + 8);
#endif
      const unsigned short* vp = Vb + (size_t)(kt * 32 + key) * EMBED + h * 128 + seg * 16;
      union { v4u u; unsigned short s[8]; } a0, a1;
      a0.u = *(const v4u*)vp;
      a1.u = *(const v4u*)(vp + 8);
      if (kt + 1 < SEQ / 32) __builtin_prefetch(vp + 32 * EMBED, 0, 1);
#pragma unroll
      for (int j = 0; j < 8; j++) {
        VT[seg * 16 + j][key]     = a0.s[j];
        VT[seg * 16 + 8 + j][key] = a1.s[j];
      }
    }
#if HAVE_TDM
    if (w == 0) __builtin_amdgcn_s_wait_tensorcnt((short)0);
#endif
    __syncthreads();

    // --- scores: S[attn][key subtile 0/1], chained K=32 over head dim 64 ---
    v8f S[2][2];
#pragma unroll
    for (int a = 0; a < 2; a++)
#pragma unroll
      for (int j = 0; j < 2; j++) {
        v8f s = vz;
#pragma unroll
        for (int c = 0; c < 2; c++) {
          Frag kf;
          int key = j * 16 + lo;
          int col = a * 64 + c * 32;
          kf.u[0] = *(const v4u*)&Kls[key][col + kh * 8];
          kf.u[1] = *(const v4u*)&Kls[key][col + 16 + kh * 8];
          s = __builtin_amdgcn_wmma_f32_16x16x32_bf16(false, qf[a][c].v, false, kf.v,
                                                      (short)0, s, false, false);
        }
        S[a][j] = s;
      }

    // --- online softmax + PV per attention stream ---
#pragma unroll
    for (int a = 0; a < 2; a++) {
#pragma unroll
      for (int r = 0; r < 8; r++) {
        float v0 = S[a][0][r] * SC, v1 = S[a][1][r] * SC;
        float t = fmaxf(v0, v1);
        t = fmaxf(t, __shfl_xor(t, 1));
        t = fmaxf(t, __shfl_xor(t, 2));
        t = fmaxf(t, __shfl_xor(t, 4));
        t = fmaxf(t, __shfl_xor(t, 8));
        float nm = fmaxf(mr[a][r], t);
        float co = __expf(mr[a][r] - nm);
        float p0 = __expf(v0 - nm), p1 = __expf(v1 - nm);
        float rs = p0 + p1;
        rs += __shfl_xor(rs, 1);
        rs += __shfl_xor(rs, 2);
        rs += __shfl_xor(rs, 4);
        rs += __shfl_xor(rs, 8);
        lr[a][r] = lr[a][r] * co + rs;
        mr[a][r] = nm;
#pragma unroll
        for (int d = 0; d < 8; d++) acc[a][d][r] = acc[a][d][r] * co;
        Pls[w][kh * 8 + r][lo]      = f2bf(p0);
        Pls[w][kh * 8 + r][16 + lo] = f2bf(p1);
      }
      Frag pf;
      pf.u[0] = *(const v4u*)&Pls[w][lo][kh * 8];
      pf.u[1] = *(const v4u*)&Pls[w][lo][16 + kh * 8];
#pragma unroll
      for (int d = 0; d < 8; d++) {
        Frag vf;
        vf.u[0] = *(const v4u*)&VT[d * 16 + lo][kh * 8];
        vf.u[1] = *(const v4u*)&VT[d * 16 + lo][16 + kh * 8];
        acc[a][d] = __builtin_amdgcn_wmma_f32_16x16x32_bf16(false, pf.v, false, vf.v,
                                                            (short)0, acc[a][d], false, false);
      }
    }
  }

  // --- epilogue: diff combine, RMS norm over 128 dims, subln, scale, store ---
  const float lam = lam_p[0];
  const float OUTS = 1.0f - LAMBDA_INIT;
  float sw[8];
#pragma unroll
  for (int d = 0; d < 8; d++) sw[d] = subln[d * 16 + lo];

#pragma unroll
  for (int r = 0; r < 8; r++) {
    float il1 = 1.0f / lr[0][r];
    float il2 = 1.0f / lr[1][r];
    float o[8], ss = 0.f;
#pragma unroll
    for (int d = 0; d < 8; d++) {
      float v = acc[0][d][r] * il1 - lam * (acc[1][d][r] * il2);
      o[d] = v;
      ss += v * v;
    }
    ss += __shfl_xor(ss, 1);
    ss += __shfl_xor(ss, 2);
    ss += __shfl_xor(ss, 4);
    ss += __shfl_xor(ss, 8);
    float rms = rsqrtf(ss * (1.0f / 128.0f) + 1e-5f);
    int trow = qb * 128 + w * 16 + kh * 8 + r;
    unsigned short* op = attnb + (size_t)trow * EMBED + h * 128;
#pragma unroll
    for (int d = 0; d < 8; d++) op[d * 16 + lo] = f2bf(o[d] * rms * sw[d] * OUTS);
  }
}

// ---------------------------------------------------------------------------
// Host-side launch
// ---------------------------------------------------------------------------
extern "C" void kernel_launch(void* const* d_in, const int* in_sizes, int n_in,
                              void* d_out, int out_size, void* d_ws, size_t ws_size,
                              hipStream_t stream) {
  (void)in_sizes; (void)n_in; (void)out_size; (void)ws_size;
  const float* x     = (const float*)d_in[0];
  const float* wq    = (const float*)d_in[1];
  const float* wk    = (const float*)d_in[2];
  const float* wv    = (const float*)d_in[3];
  const float* wo    = (const float*)d_in[4];
  const float* lq1   = (const float*)d_in[5];
  const float* lk1   = (const float*)d_in[6];
  const float* lq2   = (const float*)d_in[7];
  const float* lk2   = (const float*)d_in[8];
  const float* subln = (const float*)d_in[9];

  const size_t NE = (size_t)EMBED * SEQ;  // 4M elements
  char* ws = (char*)d_ws;
  float* lam = (float*)ws;
  size_t off = 256;
  auto nxt = [&](size_t elems) -> unsigned short* {
    unsigned short* p = (unsigned short*)(ws + off);
    off += ((elems * 2 + 255) & ~(size_t)255);
    return p;
  };
  unsigned short* xb  = nxt(NE);
  unsigned short* wqb = nxt(NE);
  unsigned short* wkb = nxt(NE);
  unsigned short* wvb = nxt(NE);
  unsigned short* wob = nxt(NE);
  unsigned short* Qb  = nxt(NE);
  unsigned short* Kb  = nxt(NE);
  unsigned short* Vb  = nxt(NE);
  unsigned short* Ab  = nxt(NE);   // total ws use: ~72 MB + lambda

  const int cvtBlocks = (int)(NE / 1024);
  cvt_bf16_kernel<<<cvtBlocks, 256, 0, stream>>>(x,  xb,  (int)NE);
  cvt_bf16_kernel<<<cvtBlocks, 256, 0, stream>>>(wq, wqb, (int)NE);
  cvt_bf16_kernel<<<cvtBlocks, 256, 0, stream>>>(wk, wkb, (int)NE);
  cvt_bf16_kernel<<<cvtBlocks, 256, 0, stream>>>(wv, wvb, (int)NE);
  cvt_bf16_kernel<<<cvtBlocks, 256, 0, stream>>>(wo, wob, (int)NE);
  lambda_kernel<<<1, 32, 0, stream>>>(lq1, lk1, lq2, lk2, lam);

  dim3 g16(16, 16);
  gemm2048_kernel<<<g16, 256, 0, stream>>>(xb, wqb, Qb, nullptr);
  gemm2048_kernel<<<g16, 256, 0, stream>>>(xb, wkb, Kb, nullptr);
  gemm2048_kernel<<<g16, 256, 0, stream>>>(xb, wvb, Vb, nullptr);

  diffattn_kernel<<<g16, 256, 0, stream>>>(Qb, Kb, Vb, lam, subln, Ab);

  gemm2048_kernel<<<g16, 256, 0, stream>>>(Ab, wob, nullptr, (float*)d_out);
}
